// GlobalGNN_9689446219793
// MI455X (gfx1250) — matrix operand order repaired and checked by reading, not compiled, not run on hardware
//
#include <hip/hip_runtime.h>
#include <hip/hip_bf16.h>

#define NMOL 128
#define NPM 48
#define DIM 128
#define NATOMS (NMOL * NPM)
#define CUTOFF_F 10.0f
#define PI_F 3.14159f
#define EPS_DIST 1e-6f
#define LN_EPS 1e-5f

// packed-fragment geometry: one fragment = 32 lanes x 16 f16 = 512 f16 (1 KB)
#define FRAG_ELEMS 512
// per-layer packed-weight fragment offsets (units of 512 f16)
#define PW_W1A 0
#define PW_W1B 32
#define PW_W1C 64
#define PW_MW2 72
#define PW_UW1A 104
#define PW_UW1B 136
#define PW_UW2 168
#define PW_DEW2 200
#define PW_LAYER_FRAGS 201

#if defined(__gfx1250__) &&                                            \
    __has_builtin(__builtin_amdgcn_global_load_async_to_lds_b128) &&   \
    __has_builtin(__builtin_amdgcn_s_wait_asynccnt)
#define USE_ASYNC_LDS 1
#else
#define USE_ASYNC_LDS 0
#endif

typedef __attribute__((ext_vector_type(16))) _Float16 v16h;
typedef __attribute__((ext_vector_type(8)))  float    v8f;
typedef int v4i __attribute__((vector_size(16)));  // builtin's operand type

__device__ __forceinline__ v8f wmma32(v16h a, v16h b, v8f c) {
  return __builtin_amdgcn_wmma_f32_16x16x32_f16(false, a, false, b, (short)0, c,
                                                false, false);
}

__device__ __forceinline__ float silu_f(float x) {
  return x / (1.0f + __expf(-x));
}

// Packed B-fragment load: lane's 16 f16 are contiguous (32B aligned).
__device__ __forceinline__ v16h load_bp(const _Float16* __restrict__ base,
                                        int kk, int nt, int lane) {
  return *(const v16h*)(base + (size_t)((kk * 8 + nt) * 32 + lane) * 16);
}

// ---------------------------------------------------------------------------
// Weight packer: row-major f32 W[K x ld] -> f16 B-fragment order.
// Fragment (kk,nt), lane l, elem e <-> W[kk*32 + (e<8 ? kb+e : 16+kb+e-8)][nt*16 + (l&15)],
// kb = (l<16)?0:8; zero-padded for k >= kmax.
// ---------------------------------------------------------------------------
__global__ void pack_b_kernel(const float* __restrict__ w, int ld, int kmax,
                              _Float16* __restrict__ dst) {
  int t = blockIdx.x * blockDim.x + threadIdx.x;  // frag*32 + lane
  int lane = t & 31;
  int frag = t >> 5;
  int kk = frag >> 3, nt = frag & 7;
  int col = nt * 16 + (lane & 15);
  int kb = (lane < 16) ? 0 : 8;
  v16h b;
#pragma unroll
  for (int e = 0; e < 16; ++e) {
    int k = kk * 32 + ((e < 8) ? (kb + e) : (16 + kb + e - 8));
    b[e] = (k < kmax) ? (_Float16)w[k * ld + col] : (_Float16)0.0f;
  }
  *(v16h*)(dst + (size_t)t * 16) = b;
}

// ---------------------------------------------------------------------------
// Activation packer: row-major f32 h[N x 128] -> f16 A-fragment order.
// A layout (ISA 7.12.2): lane<16 row=lane holds K 0..7,16..23;
// lane>=16 row=lane-16 holds K 8..15,24..31.
// ---------------------------------------------------------------------------
__global__ void pack_a_kernel(const float* __restrict__ h,
                              _Float16* __restrict__ dst) {
  int t = blockIdx.x * blockDim.x + threadIdx.x;  // (tile*4+kk)*32 + lane
  int lane = t & 31;
  int fk = t >> 5;
  int kk = fk & 3;
  int tile = fk >> 2;
  int row = lane & 15;
  int kb = (lane < 16) ? 0 : 8;
  v16h a;
#pragma unroll
  for (int e = 0; e < 16; ++e) {
    int v = e >> 1, p = e & 1;
    int k = (v < 4) ? (kb + 2 * v + p) : (16 + kb + 2 * (v - 4) + p);
    a[e] = (_Float16)h[(size_t)(tile * 16 + row) * DIM + kk * 32 + k];
  }
  *(v16h*)(dst + (size_t)t * 16) = a;
}

// ---------------------------------------------------------------------------
// Kernel 1: per-atom projections (SA = h@W1a, SB = h@W1b + m_b1,
// U = h@u_w1a + u_b1). One wave per 16-atom tile. Per n-tile all 12 B
// fragments are loaded as one batch so the loads overlap the WMMAs.
// ---------------------------------------------------------------------------
__global__ void atom_proj_kernel(const _Float16* __restrict__ hA,
                                 const _Float16* __restrict__ pw,
                                 const float* __restrict__ m_b1,
                                 const float* __restrict__ u_b1,
                                 float* __restrict__ SA,
                                 float* __restrict__ SB,
                                 float* __restrict__ U) {
  int gwave = (blockIdx.x * blockDim.x + threadIdx.x) >> 5;
  int lane = threadIdx.x & 31;
  int row0 = gwave * 16;
  if (row0 >= NATOMS) return;  // wave-uniform

  __builtin_prefetch(pw, 0, 3);

  v16h af[4];
#pragma unroll
  for (int kk = 0; kk < 4; ++kk)
    af[kk] = *(const v16h*)(hA + (size_t)((gwave * 4 + kk) * 32 + lane) * 16);

  const _Float16* bA = pw + (size_t)PW_W1A * FRAG_ELEMS;
  const _Float16* bB = pw + (size_t)PW_W1B * FRAG_ELEMS;
  const _Float16* bU = pw + (size_t)PW_UW1A * FRAG_ELEMS;

  int n = lane & 15;
#pragma unroll
  for (int nt = 0; nt < 8; ++nt) {
    v16h bfA[4], bfB[4], bfU[4];
#pragma unroll
    for (int kk = 0; kk < 4; ++kk) {  // 24x b128 issued before first WMMA
      bfA[kk] = load_bp(bA, kk, nt, lane);
      bfB[kk] = load_bp(bB, kk, nt, lane);
      bfU[kk] = load_bp(bU, kk, nt, lane);
    }
    v8f aA = {};
    v8f aB = {};
    v8f aU = {};
#pragma unroll
    for (int kk = 0; kk < 4; ++kk) {
      aA = wmma32(af[kk], bfA[kk], aA);
      aB = wmma32(af[kk], bfB[kk], aB);
      aU = wmma32(af[kk], bfU[kk], aU);
    }
    int col = nt * 16 + n;
#pragma unroll
    for (int r = 0; r < 8; ++r) {
      int row = row0 + ((lane < 16) ? r : 8 + r);
      SA[row * DIM + col] = aA[r];
      SB[row * DIM + col] = aB[r] + m_b1[col];
      U[row * DIM + col] = aU[r] + u_b1[col];
    }
  }
}

// ---------------------------------------------------------------------------
// Kernel 2: one block per molecule, 3 waves; wave w owns dst tile [16w,16w+16).
// The dist-embed second layer is a WMMA too (K=16 padded): t-tile -> de_w2,
// then (D + de_b2)*cw is scattered into the feat A-tile for the W1c WMMA.
// ---------------------------------------------------------------------------
__global__ void pair_update_kernel(const float* __restrict__ h_in,
                                   const float* __restrict__ pos,
                                   const float* __restrict__ SA,
                                   const float* __restrict__ SB,
                                   const float* __restrict__ U,
                                   const _Float16* __restrict__ pw,
                                   const float* __restrict__ m_b2,
                                   const float* __restrict__ u_b2,
                                   const float* __restrict__ de_w1,
                                   const float* __restrict__ de_b1,
                                   const float* __restrict__ de_b2,
                                   const float* __restrict__ ln_g,
                                   const float* __restrict__ ln_b,
                                   float* __restrict__ h_out) {
  __shared__ float sPos[NPM * 3];
  __shared__ __align__(16) float sSA[NPM * DIM];
  __shared__ float sDE[48];    // de_w1(16) de_b1(16) de_b2(16)
  __shared__ float sCons[512]; // m_b2, u_b2, ln_g, ln_b
  __shared__ float sMask[48];
  __shared__ float sCw[48];
  __shared__ float sCnt[48];
  __shared__ __align__(32) _Float16 sF16[3 * 32 * 16];      // t A-tiles
  __shared__ __align__(32) _Float16 sG16[3 * 32 * 16];      // feat A-tiles
  __shared__ __align__(32) _Float16 sT16[3 * 4 * 32 * 16];  // staging A-tiles

  int mol = blockIdx.x;
  int tid = threadIdx.x;
  int wave = tid >> 5;
  int lane = tid & 31;
  int atom0 = mol * NPM;

  __builtin_prefetch(pw + (size_t)PW_W1C * FRAG_ELEMS, 0, 3);

  // ---- phase 0: cooperative staging ----
#if USE_ASYNC_LDS
  {
    typedef __attribute__((address_space(1))) v4i* gv4i_p;
    typedef __attribute__((address_space(3))) v4i* lv4i_p;
    gv4i_p gsa =
        (gv4i_p)(unsigned long long)(size_t)(SA + (size_t)atom0 * DIM);
    lv4i_p lsa = (lv4i_p)(unsigned int)(size_t)(float*)sSA;
    for (int c = tid; c < NPM * DIM / 4; c += blockDim.x)
      __builtin_amdgcn_global_load_async_to_lds_b128(gsa + c, lsa + c, 0, 0);
  }
#else
  for (int idx = tid; idx < NPM * DIM; idx += blockDim.x)
    sSA[idx] = SA[atom0 * DIM + idx];
#endif
  for (int idx = tid; idx < NPM * 3; idx += blockDim.x)
    sPos[idx] = pos[atom0 * 3 + idx];
  for (int idx = tid; idx < 16; idx += blockDim.x) {
    sDE[idx] = de_w1[idx];
    sDE[16 + idx] = de_b1[idx];
    sDE[32 + idx] = de_b2[idx];
  }
  for (int idx = tid; idx < 128; idx += blockDim.x) {
    sCons[idx] = m_b2[idx];
    sCons[128 + idx] = u_b2[idx];
    sCons[256 + idx] = ln_g[idx];
    sCons[384 + idx] = ln_b[idx];
  }
  for (int idx = tid; idx < 3 * 32 * 16; idx += blockDim.x) {
    sF16[idx] = (_Float16)0.0f;  // K>=16 padding stays zero forever
    sG16[idx] = (_Float16)0.0f;
  }
#if USE_ASYNC_LDS
  __builtin_amdgcn_s_wait_asynccnt(0);
#endif
  __syncthreads();

  int jt = wave;
  int n = lane & 15;
  _Float16* sF16w = sF16 + wave * 512;
  _Float16* sG16w = sG16 + wave * 512;
  _Float16* sT16w = sT16 + wave * 2048;
  const _Float16* bW1c = pw + (size_t)PW_W1C * FRAG_ELEMS;
  const _Float16* bMW2 = pw + (size_t)PW_MW2 * FRAG_ELEMS;
  const _Float16* bUW1b = pw + (size_t)PW_UW1B * FRAG_ELEMS;
  const _Float16* bUW2 = pw + (size_t)PW_UW2 * FRAG_ELEMS;
  v16h bDE = load_bp(pw + (size_t)PW_DEW2 * FRAG_ELEMS, 0, 0, lane);

  // Hoist loop-invariant SB addend (dest projection, m_b1 folded).
  float sbv[8][8];
#pragma unroll
  for (int nt = 0; nt < 8; ++nt)
#pragma unroll
    for (int r = 0; r < 8; ++r) {
      int jl = (lane < 16) ? r : (8 + r);
      sbv[nt][r] = SB[(size_t)(atom0 + jt * 16 + jl) * DIM + nt * 16 + n];
    }

  v8f aggAcc[8];
#pragma unroll
  for (int nt = 0; nt < 8; ++nt) aggAcc[nt] = (v8f){};
  float cntLocal = 0.0f;

  // ---- phase 1: loop over source atoms ----
  for (int i = 0; i < NPM; ++i) {
    if (lane < 16) {
      int j = jt * 16 + lane;
      float dx = sPos[i * 3 + 0] - sPos[j * 3 + 0];
      float dy = sPos[i * 3 + 1] - sPos[j * 3 + 1];
      float dz = sPos[i * 3 + 2] - sPos[j * 3 + 2];
      float dist = fmaxf(sqrtf(dx * dx + dy * dy + dz * dz), EPS_DIST);
      float maskf = ((dist < CUTOFF_F) && (i != j)) ? 1.0f : 0.0f;
      float cw = 0.5f * (1.0f + __cosf(PI_F * dist / CUTOFF_F));
#pragma unroll
      for (int k = 0; k < 16; ++k) {
        float t = silu_f(dist * sDE[k] + sDE[16 + k]);
        int lane_t = lane + (((k >> 3) & 1) << 4);
        sF16w[lane_t * 16 + (k & 7)] = (_Float16)t;
      }
      sMask[jt * 16 + lane] = maskf;
      sCw[jt * 16 + lane] = cw;
      cntLocal += maskf;
    }
    __syncthreads();

    // dist-embed layer 2 on the matrix pipe (16 pairs x 16 x 16, K padded)
    v16h at = *(const v16h*)(sF16w + lane * 16);
    v8f z0 = {};
    v8f fD = wmma32(at, bDE, z0);
#pragma unroll
    for (int r = 0; r < 8; ++r) {
      int jl = (lane < 16) ? r : (8 + r);
      float f = (fD[r] + sDE[32 + n]) * sCw[jt * 16 + jl];
      int lane_t = jl + (((n >> 3) & 1) << 4);
      sG16w[lane_t * 16 + (n & 7)] = (_Float16)f;
    }
    __syncthreads();

    v16h bf[8];
#pragma unroll
    for (int nt = 0; nt < 8; ++nt) bf[nt] = load_bp(bW1c, 0, nt, lane);
    v16h afeat = *(const v16h*)(sG16w + lane * 16);
#pragma unroll
    for (int nt = 0; nt < 8; ++nt) {
      v8f z = {};
      v8f acc = wmma32(afeat, bf[nt], z);
      float sa = sSA[i * DIM + nt * 16 + n];
#pragma unroll
      for (int r = 0; r < 8; ++r) {
        int jl = (lane < 16) ? r : (8 + r);
        float pre = acc[r] + sa + sbv[nt][r];
        aggAcc[nt][r] += sMask[jt * 16 + jl] * silu_f(pre);
      }
    }
    __syncthreads();
  }

  // ---- phase 2 ----
  auto store_tile16 = [&](const v8f* t) {
#pragma unroll
    for (int nt = 0; nt < 8; ++nt)
#pragma unroll
      for (int r = 0; r < 8; ++r) {
        int jl = (lane < 16) ? r : (8 + r);
        int kk = nt >> 1;
        int lane_t = jl + (((n >> 3) & 1) << 4);
        int e = (n & 7) + ((nt & 1) << 3);
        sT16w[(kk * 32 + lane_t) * 16 + e] = (_Float16)t[nt][r];
      }
  };
  auto load_afrags = [&](v16h* af) {
#pragma unroll
    for (int kk = 0; kk < 4; ++kk)
      af[kk] = *(const v16h*)(sT16w + (kk * 32 + lane) * 16);
  };

  store_tile16(aggAcc);
  if (lane < 16) sCnt[jt * 16 + lane] = cntLocal;
  __syncthreads();

  v16h af[4];
  v8f tv[8];

  // agg = aggpre @ m_w2 + cnt * m_b2
  load_afrags(af);
#pragma unroll
  for (int nt = 0; nt < 8; ++nt) {
    v16h bf[4];
#pragma unroll
    for (int kk = 0; kk < 4; ++kk) bf[kk] = load_bp(bMW2, kk, nt, lane);
    v8f acc = {};
#pragma unroll
    for (int kk = 0; kk < 4; ++kk) acc = wmma32(af[kk], bf[kk], acc);
    int hid = nt * 16 + n;
#pragma unroll
    for (int r = 0; r < 8; ++r) {
      int jl = (lane < 16) ? r : (8 + r);
      tv[nt][r] = acc[r] + sCnt[jt * 16 + jl] * sCons[hid];
    }
  }
  __syncthreads();
  store_tile16(tv);
  __syncthreads();

  // x1 = silu(U + agg @ u_w1[128:256,:])
  load_afrags(af);
#pragma unroll
  for (int nt = 0; nt < 8; ++nt) {
    v16h bf[4];
#pragma unroll
    for (int kk = 0; kk < 4; ++kk) bf[kk] = load_bp(bUW1b, kk, nt, lane);
    v8f acc = {};
#pragma unroll
    for (int kk = 0; kk < 4; ++kk) acc = wmma32(af[kk], bf[kk], acc);
    int hid = nt * 16 + n;
#pragma unroll
    for (int r = 0; r < 8; ++r) {
      int jl = (lane < 16) ? r : (8 + r);
      tv[nt][r] = silu_f(acc[r] + U[(size_t)(atom0 + jt * 16 + jl) * DIM + hid]);
    }
  }
  __syncthreads();
  store_tile16(tv);
  __syncthreads();

  // x = h_in + x1 @ u_w2 + u_b2
  load_afrags(af);
#pragma unroll
  for (int nt = 0; nt < 8; ++nt) {
    v16h bf[4];
#pragma unroll
    for (int kk = 0; kk < 4; ++kk) bf[kk] = load_bp(bUW2, kk, nt, lane);
    v8f acc = {};
#pragma unroll
    for (int kk = 0; kk < 4; ++kk) acc = wmma32(af[kk], bf[kk], acc);
    int hid = nt * 16 + n;
#pragma unroll
    for (int r = 0; r < 8; ++r) {
      int jl = (lane < 16) ? r : (8 + r);
      tv[nt][r] = acc[r] + sCons[128 + hid] +
                  h_in[(size_t)(atom0 + jt * 16 + jl) * DIM + hid];
    }
  }

  // LayerNorm fully in registers: 16-lane shuffle reductions per row half.
  float mu[8], inv[8];
#pragma unroll
  for (int r = 0; r < 8; ++r) {
    float s = 0.0f;
#pragma unroll
    for (int nt = 0; nt < 8; ++nt) s += tv[nt][r];
    s += __shfl_xor(s, 1, 32);
    s += __shfl_xor(s, 2, 32);
    s += __shfl_xor(s, 4, 32);
    s += __shfl_xor(s, 8, 32);
    mu[r] = s * (1.0f / DIM);
  }
#pragma unroll
  for (int r = 0; r < 8; ++r) {
    float s = 0.0f;
#pragma unroll
    for (int nt = 0; nt < 8; ++nt) {
      float d = tv[nt][r] - mu[r];
      s += d * d;
    }
    s += __shfl_xor(s, 1, 32);
    s += __shfl_xor(s, 2, 32);
    s += __shfl_xor(s, 4, 32);
    s += __shfl_xor(s, 8, 32);
    inv[r] = rsqrtf(s * (1.0f / DIM) + LN_EPS);
  }
#pragma unroll
  for (int nt = 0; nt < 8; ++nt) {
    int hid = nt * 16 + n;
#pragma unroll
    for (int r = 0; r < 8; ++r) {
      int jl = (lane < 16) ? r : (8 + r);
      h_out[(size_t)(atom0 + jt * 16 + jl) * DIM + hid] =
          (tv[nt][r] - mu[r]) * inv[r] * sCons[256 + hid] + sCons[384 + hid];
    }
  }
}

// ---------------------------------------------------------------------------
extern "C" void kernel_launch(void* const* d_in, const int* in_sizes, int n_in,
                              void* d_out, int out_size, void* d_ws,
                              size_t ws_size, hipStream_t stream) {
  (void)in_sizes; (void)n_in; (void)out_size; (void)ws_size;

  const float* h0 = (const float*)d_in[0];
  const float* pos = (const float*)d_in[1];
  // d_in[2] batch_idx encodes equal-sized groups; unused.

  float* ws = (float*)d_ws;
  float* SA = ws;
  float* SB = SA + (size_t)NATOMS * DIM;
  float* Ub = SB + (size_t)NATOMS * DIM;
  float* hT = Ub + (size_t)NATOMS * DIM;
  _Float16* hA = (_Float16*)(hT + (size_t)NATOMS * DIM);  // packed h, 1.5 MB
  _Float16* pw[2];
  pw[0] = hA + (size_t)NATOMS * DIM;
  pw[1] = pw[0] + (size_t)PW_LAYER_FRAGS * FRAG_ELEMS;

  // ---- pack all weights (both layers) into f16 fragment order ----
  for (int l = 0; l < 2; ++l) {
    const float* de_w2 = (const float*)d_in[5] + l * 256;
    const float* m_w1 = (const float*)d_in[7] + l * 272 * 128;
    const float* m_w2 = (const float*)d_in[9] + l * 128 * 128;
    const float* u_w1 = (const float*)d_in[11] + l * 256 * 128;
    const float* u_w2 = (const float*)d_in[13] + l * 128 * 128;
    pack_b_kernel<<<16, 64, 0, stream>>>(m_w1, DIM, 128,
                                         pw[l] + (size_t)PW_W1A * FRAG_ELEMS);
    pack_b_kernel<<<16, 64, 0, stream>>>(m_w1 + 128 * 128, DIM, 128,
                                         pw[l] + (size_t)PW_W1B * FRAG_ELEMS);
    pack_b_kernel<<<4, 64, 0, stream>>>(m_w1 + 256 * 128, DIM, 16,
                                        pw[l] + (size_t)PW_W1C * FRAG_ELEMS);
    pack_b_kernel<<<16, 64, 0, stream>>>(m_w2, DIM, 128,
                                         pw[l] + (size_t)PW_MW2 * FRAG_ELEMS);
    pack_b_kernel<<<16, 64, 0, stream>>>(u_w1, DIM, 128,
                                         pw[l] + (size_t)PW_UW1A * FRAG_ELEMS);
    pack_b_kernel<<<16, 64, 0, stream>>>(u_w1 + 128 * 128, DIM, 128,
                                         pw[l] + (size_t)PW_UW1B * FRAG_ELEMS);
    pack_b_kernel<<<16, 64, 0, stream>>>(u_w2, DIM, 128,
                                         pw[l] + (size_t)PW_UW2 * FRAG_ELEMS);
    pack_b_kernel<<<1, 32, 0, stream>>>(de_w2, 16, 16,
                                        pw[l] + (size_t)PW_DEW2 * FRAG_ELEMS);
  }

  const float* hcur = h0;
  for (int l = 0; l < 2; ++l) {
    const float* de_w1 = (const float*)d_in[3] + l * 16;
    const float* de_b1 = (const float*)d_in[4] + l * 16;
    const float* de_b2 = (const float*)d_in[6] + l * 16;
    const float* m_b1 = (const float*)d_in[8] + l * 128;
    const float* m_b2 = (const float*)d_in[10] + l * 128;
    const float* u_b1 = (const float*)d_in[12] + l * 128;
    const float* u_b2 = (const float*)d_in[14] + l * 128;
    const float* ln_g = (const float*)d_in[15] + l * 128;
    const float* ln_b = (const float*)d_in[16] + l * 128;

    pack_a_kernel<<<192, 256, 0, stream>>>(hcur, hA);

    atom_proj_kernel<<<96, 128, 0, stream>>>(hA, pw[l], m_b1, u_b1, SA, SB, Ub);

    float* outp = (l == 1) ? (float*)d_out : hT;
    pair_update_kernel<<<NMOL, 96, 0, stream>>>(
        hcur, pos, SA, SB, Ub, pw[l], m_b2, u_b2, de_w1, de_b1, de_b2, ln_g,
        ln_b, outp);
    hcur = outp;
  }
}
